// CausalTemporalAttention_76029511074405
// MI455X (gfx1250) — matrix operand order repaired and checked by reading, not compiled, not run on hardware
//
#include <hip/hip_runtime.h>
#include <stdint.h>

#if __has_builtin(__builtin_amdgcn_tensor_load_to_lds) && \
    __has_builtin(__builtin_amdgcn_tensor_store_from_lds) && \
    __has_builtin(__builtin_amdgcn_s_wait_tensorcnt)
#define HAVE_TDM 1
#else
#define HAVE_TDM 0
#endif

// ---------------- types / helpers ----------------
typedef __bf16 bf16_t;
typedef bf16_t v16bf __attribute__((ext_vector_type(16)));
typedef float  v8f   __attribute__((ext_vector_type(8)));
typedef float  f32x4 __attribute__((ext_vector_type(4)));
typedef unsigned int u32x4 __attribute__((ext_vector_type(4)));
typedef unsigned int u32x2 __attribute__((ext_vector_type(2)));
typedef int i32x8 __attribute__((ext_vector_type(8)));
typedef int i32x4 __attribute__((ext_vector_type(4)));

#define C_DIM 512
#define T_DIM 32
#define B_DIM 4
#define HW    1024
#define P_DIM 4096                 // B*H*W
#define N_COLS (P_DIM * T_DIM)     // 131072
#define GN_EPS 1e-6f

__device__ __forceinline__ unsigned short f2bf(float f) {
    unsigned u = __float_as_uint(f);
    u += 0x7FFFu + ((u >> 16) & 1u);     // round-to-nearest-even
    return (unsigned short)(u >> 16);
}
__device__ __forceinline__ float bf2f(unsigned h) {
    return __uint_as_float(h << 16);
}
__device__ __forceinline__ unsigned pack_bf2(float lo, float hi) {
    return (unsigned)f2bf(lo) | ((unsigned)f2bf(hi) << 16);
}

union Frag { v16bf v; u32x4 q[2]; };

#if HAVE_TDM
// Build D# groups for a 2-D tile (2-byte elements): X contiguous, Y rows strided.
// padIntervalCode: 0=2,1=4,2=8,3=16,4=32,5=64 DWORDs; padAmountCode: N-1 DWORDs of pad.
__device__ __forceinline__ void tdm_build(u32x4& g0, i32x8& g1,
                                          const void* lds, const void* gmem,
                                          unsigned tileX, unsigned tileY,
                                          unsigned tensorX, unsigned tensorY,
                                          unsigned long long strideX,
                                          int padEn, unsigned padI, unsigned padA) {
    const unsigned long long ga = (unsigned long long)gmem;
    g0[0] = 1u;                                                  // count=1, user, no gather
    g0[1] = (unsigned)(unsigned long long)lds;                   // lds_addr (low 32 bits)
    g0[2] = (unsigned)ga;
    g0[3] = (unsigned)((ga >> 32) & 0x01FFFFFFu) | (2u << 30);   // addr[56:32] | type=2
    g1[0] = (int)((1u << 16) | ((unsigned)(padEn ? 1u : 0u) << 20) |
                  (padI << 22) | (padA << 25));                  // data_size=2B, pad ctl
    g1[1] = (int)((tensorX & 0xFFFFu) << 16);                    // tensor_dim0[15:0] @48
    g1[2] = (int)(((tensorX >> 16) & 0xFFFFu) | ((tensorY & 0xFFFFu) << 16));
    g1[3] = (int)(((tensorY >> 16) & 0xFFFFu) | ((tileX & 0xFFFFu) << 16));
    g1[4] = (int)(tileY & 0xFFFFu);                              // tile_dim1
    g1[5] = (int)(unsigned)(strideX & 0xFFFFFFFFull);            // dim0_stride @160
    g1[6] = (int)(unsigned)((strideX >> 32) & 0xFFFFull);
    g1[7] = 0;
}
__device__ __forceinline__ void tdm_load_tile_2d(const void* lds, const void* gsrc,
                                                 unsigned tileX, unsigned tileY,
                                                 unsigned tensorX, unsigned tensorY,
                                                 unsigned long long strideX,
                                                 int padEn, unsigned padI, unsigned padA) {
    u32x4 g0; i32x8 g1;
    tdm_build(g0, g1, lds, gsrc, tileX, tileY, tensorX, tensorY, strideX, padEn, padI, padA);
    i32x4 z4 = {0, 0, 0, 0};
#if __clang_major__ >= 23
    i32x8 z8 = {0, 0, 0, 0, 0, 0, 0, 0};
    __builtin_amdgcn_tensor_load_to_lds(g0, g1, z4, z4, z8, 0);
#else
    __builtin_amdgcn_tensor_load_to_lds(g0, g1, z4, z4, 0);
#endif
}
__device__ __forceinline__ void tdm_store_tile_2d(const void* lds, void* gdst,
                                                  unsigned tileX, unsigned tileY,
                                                  unsigned tensorX, unsigned tensorY,
                                                  unsigned long long strideX) {
    u32x4 g0; i32x8 g1;
    tdm_build(g0, g1, lds, gdst, tileX, tileY, tensorX, tensorY, strideX, 0, 0, 0);
    i32x4 z4 = {0, 0, 0, 0};
#if __clang_major__ >= 23
    i32x8 z8 = {0, 0, 0, 0, 0, 0, 0, 0};
    __builtin_amdgcn_tensor_store_from_lds(g0, g1, z4, z4, z8, 0);
#else
    __builtin_amdgcn_tensor_store_from_lds(g0, g1, z4, z4, 0);
#endif
}
#endif

// ---------------- kernel 0: convert 4 weight matrices f32 -> bf16 (once) ----------------
__global__ void __launch_bounds__(256) convert_w_kernel(const float* __restrict__ w0,
                                                        const float* __restrict__ w1,
                                                        const float* __restrict__ w2,
                                                        const float* __restrict__ w3,
                                                        unsigned short* __restrict__ Wbf) {
    const int idx = blockIdx.x * 256 + threadIdx.x;   // 0 .. 262143
    const int m = idx >> 16;
    const int e = (idx & 0xFFFF) * 4;
    const float* src = (m == 0) ? w0 : (m == 1) ? w1 : (m == 2) ? w2 : w3;
    const f32x4 v = *(const f32x4*)&src[e];
    u32x2 pk;
    pk[0] = pack_bf2(v[0], v[1]);
    pk[1] = pack_bf2(v[2], v[3]);
    *(u32x2*)&Wbf[(size_t)m * 262144 + e] = pk;
}

// ---------------- kernel 1: GroupNorm(1 group) stats per (b,t) ----------------
__global__ void __launch_bounds__(256) gn_stats_kernel(const float* __restrict__ x,
                                                       float* __restrict__ stats) {
    const int bt = blockIdx.x;            // b*32 + t
    const int b = bt >> 5, t = bt & 31;
    float s1 = 0.f, s2 = 0.f;
    for (int i = threadIdx.x; i < (C_DIM * HW) / 4; i += 256) {
        const int c = i >> 8;
        const int hw = (i & 255) * 4;
        const f32x4 v = *(const f32x4*)&x[((((size_t)b * C_DIM + c) * T_DIM) + t) * HW + hw];
        s1 += v[0] + v[1] + v[2] + v[3];
        s2 += v[0] * v[0] + v[1] * v[1] + v[2] * v[2] + v[3] * v[3];
    }
    __shared__ float r1[256], r2[256];
    r1[threadIdx.x] = s1; r2[threadIdx.x] = s2;
    __syncthreads();
    for (int off = 128; off > 0; off >>= 1) {
        if (threadIdx.x < off) {
            r1[threadIdx.x] += r1[threadIdx.x + off];
            r2[threadIdx.x] += r2[threadIdx.x + off];
        }
        __syncthreads();
    }
    if (threadIdx.x == 0) {
        const float inv = 1.f / (float)(C_DIM * HW);
        const float mean = r1[0] * inv;
        const float var  = r2[0] * inv - mean * mean;
        stats[bt * 2]     = mean;
        stats[bt * 2 + 1] = rsqrtf(var + GN_EPS);
    }
}

// ------- kernel 2: normalize + transpose -> HT[n][c] (bf16), n = (b*1024+y*32+w)*32+t -------
__global__ void __launch_bounds__(256) norm_ht_kernel(const float* __restrict__ x,
                                                      const float* __restrict__ gamma,
                                                      const float* __restrict__ beta,
                                                      const float* __restrict__ stats,
                                                      unsigned short* __restrict__ HT) {
    const int blk = blockIdx.x;           // (((b*32 + y)*32 + t)*8 + ct)
    const int ct = blk & 7;
    const int t  = (blk >> 3) & 31;
    const int y  = (blk >> 8) & 31;
    const int b  = blk >> 13;
    const int c0 = ct * 64;
    __shared__ float tile[64][36];        // [c_local][w], 144B rows
    const float mean = stats[(b * T_DIM + t) * 2];
    const float rstd = stats[(b * T_DIM + t) * 2 + 1];
    {
        #pragma unroll
        for (int i = 0; i < 2; ++i) {
            const int e = threadIdx.x + i * 256;  // 512 chunks of 4
            const int cl = e >> 3, w0 = (e & 7) * 4;
            const float g = gamma[c0 + cl], bb = beta[c0 + cl];
            const f32x4 v = *(const f32x4*)
                &x[(((size_t)(b * C_DIM + c0 + cl) * T_DIM + t) * 32 + y) * 32 + w0];
            f32x4 o;
            #pragma unroll
            for (int j = 0; j < 4; ++j) o[j] = (v[j] - mean) * rstd * g + bb;
            *(f32x4*)&tile[cl][w0] = o;
        }
    }
    __syncthreads();
    {
        const int w = threadIdx.x >> 3, cl0 = (threadIdx.x & 7) * 8;
        u32x4 pk;
        #pragma unroll
        for (int j = 0; j < 4; ++j)
            pk[j] = pack_bf2(tile[cl0 + 2 * j][w], tile[cl0 + 2 * j + 1][w]);
        const size_t n = ((size_t)(b * HW + y * 32 + w)) * T_DIM + t;
        *(u32x4*)&HT[n * C_DIM + c0 + cl0] = pk;
    }
}

// ------- kernels 3/5: NT-GEMM  D[m,n] = sum_k A[m,k]*B[n,k], K=512, both row-major(ld=512)
// Double-buffered TDM pipeline: prefetch k-step i+1 while computing i.
__global__ void __launch_bounds__(256) gemm_nt_kernel(const unsigned short* __restrict__ Amat,
                                                      const unsigned short* __restrict__ Bmat,
                                                      unsigned short* __restrict__ Out,
                                                      size_t ldOut) {
    __shared__ __align__(16) unsigned char smem[30720];   // 2 x (As 10240 + Bs 5120)
    const int tid  = threadIdx.x;
    const int lane = tid & 31;
    const int wv   = tid >> 5;
    const int wvu  = __builtin_amdgcn_readfirstlane(wv);  // scalar wave id (uniform branch)
    const int wm   = wv >> 1;
    const int wn   = wv & 1;
    const size_t m0 = (size_t)blockIdx.y * 128;
    const size_t n0 = (size_t)blockIdx.x * 64;
    const int halfsel = (lane < 16) ? 0 : 1;
    const int lrow = lane & 15;

    v8f zero = {};
    v8f acc[2][2] = {{zero, zero}, {zero, zero}};

#if HAVE_TDM
    // prologue: fetch k-step 0 into set 0 (A rows padded to 80B for conflict-free frags)
    if (wvu == 0)
        tdm_load_tile_2d(smem, &Amat[m0 * 512], 32, 128, 512, 1u << 20, 512ull, 1, 3, 3);
    if (wvu == 1)
        tdm_load_tile_2d(smem + 10240, &Bmat[n0 * 512], 32, 64, 512, 1u << 20, 512ull, 1, 3, 3);
#endif
    for (int it = 0; it < 16; ++it) {
        const int k0 = it * 32;
        const int cur = it & 1;
        unsigned short* Asc = (unsigned short*)(smem + cur * 15360);
        unsigned short* Bsc = (unsigned short*)(smem + cur * 15360 + 10240);
#if HAVE_TDM
        unsigned char* nx = smem + (cur ^ 1) * 15360;
        if (it < 15) {
            if (wvu == 0)
                tdm_load_tile_2d(nx, &Amat[m0 * 512 + k0 + 32], 32, 128, 512, 1u << 20,
                                 512ull, 1, 3, 3);
            if (wvu == 1)
                tdm_load_tile_2d(nx + 10240, &Bmat[n0 * 512 + k0 + 32], 32, 64, 512, 1u << 20,
                                 512ull, 1, 3, 3);
            if (wvu < 2) __builtin_amdgcn_s_wait_tensorcnt(1);   // current tile landed
        } else {
            if (wvu < 2) __builtin_amdgcn_s_wait_tensorcnt(0);
        }
        __syncthreads();
#else
        __syncthreads();
        #pragma unroll
        for (int i = 0; i < 2; ++i) {
            const int idx = tid + i * 256;
            const int r = idx >> 2, kk = (idx & 3) * 8;
            *(u32x4*)&Asc[r * 40 + kk] = *(const u32x4*)&Amat[(m0 + r) * 512 + k0 + kk];
        }
        {
            const int r = tid >> 2, kk = (tid & 3) * 8;
            *(u32x4*)&Bsc[r * 40 + kk] = *(const u32x4*)&Bmat[(n0 + r) * 512 + k0 + kk];
        }
        __syncthreads();
#endif
        Frag a[2], bfr[2];
        #pragma unroll
        for (int mt = 0; mt < 2; ++mt) {
            const int m = wm * 32 + mt * 16 + lrow;
            const int ka = halfsel * 8;
            a[mt].q[0] = *(const u32x4*)&Asc[m * 40 + ka];
            a[mt].q[1] = *(const u32x4*)&Asc[m * 40 + ka + 16];
        }
        #pragma unroll
        for (int nt = 0; nt < 2; ++nt) {
            const int n = wn * 32 + nt * 16 + lrow;
            const int kb = halfsel * 8;
            bfr[nt].q[0] = *(const u32x4*)&Bsc[n * 40 + kb];
            bfr[nt].q[1] = *(const u32x4*)&Bsc[n * 40 + kb + 16];
        }
        #pragma unroll
        for (int mt = 0; mt < 2; ++mt)
            #pragma unroll
            for (int nt = 0; nt < 2; ++nt)
                acc[mt][nt] = __builtin_amdgcn_wmma_f32_16x16x32_bf16(
                    false, a[mt].v, false, bfr[nt].v, (short)0, acc[mt][nt], false, false);
        __syncthreads();    // all reads of this set done before it is refilled
    }
    // epilogue: accumulators -> LDS tile [128][64] -> TDM store (or b128 fallback)
    unsigned short* Cs = (unsigned short*)smem;
    #pragma unroll
    for (int mt = 0; mt < 2; ++mt)
        #pragma unroll
        for (int nt = 0; nt < 2; ++nt) {
            const int nl = wn * 32 + nt * 16 + lrow;
            #pragma unroll
            for (int r = 0; r < 8; ++r) {
                const int ml = wm * 32 + mt * 16 + r + halfsel * 8;
                Cs[ml * 64 + nl] = f2bf(acc[mt][nt][r]);
            }
        }
    __syncthreads();
#if HAVE_TDM
    if (wvu == 0)
        tdm_store_tile_2d(Cs, &Out[m0 * ldOut + n0], 64, 128, 1u << 20, 1u << 20,
                          (unsigned long long)ldOut);
    // s_endpgm's implicit wait-idle drains TENSORcnt
#else
    #pragma unroll
    for (int i = 0; i < 4; ++i) {
        const int e = (tid + i * 256) * 8;
        const int ml = e >> 6, nl = e & 63;
        *(u32x4*)&Out[(m0 + ml) * ldOut + n0 + nl] = *(const u32x4*)&Cs[e];
    }
#endif
}

// ------- kernel 4: per-pixel causal attention, T=32, C=512 (block = 1 pixel, 8 waves) -------
// QT/KT are [n][c] (ld=512); V is [c][n] (ld=N_COLS); output OT is [n][c] (ld=512).
// LDS map: [0..34816)  q/k double-buffered sets (S-phase), aliased by v sets (O-phase)
//          [34816..67584) sred, [67584..70144) pbf
__global__ void __launch_bounds__(256) attention_kernel(const unsigned short* __restrict__ QT,
                                                        const unsigned short* __restrict__ KT,
                                                        const unsigned short* __restrict__ Vb,
                                                        unsigned short* __restrict__ OT) {
    __shared__ __align__(16) unsigned char smem[70144];
    const int p    = blockIdx.x;
    const int tid  = threadIdx.x;
    const int lane = tid & 31;
    const int wv   = tid >> 5;
    const int wvu  = __builtin_amdgcn_readfirstlane(wv);
    const int halfsel = (lane < 16) ? 0 : 1;
    const int lrow = lane & 15;
    const size_t colbase = (size_t)p * T_DIM;
    float* sred = (float*)(smem + 34816);                 // [8][32][32]
    unsigned short* pbf = (unsigned short*)(smem + 67584);// [32][40] padded

    v8f zero = {};
    v8f Sacc[2][2] = {{zero, zero}, {zero, zero}};

    // ---- S-phase: double-buffered q/k chunk pipeline (4 chunks of 128 channels) ----
#if HAVE_TDM
    if (wvu == 0)
        tdm_load_tile_2d(smem, &QT[colbase * 512], 128, 32, 512, 1u << 20, 512ull, 1, 5, 3);
    if (wvu == 1)
        tdm_load_tile_2d(smem + 8704, &KT[colbase * 512], 128, 32, 512, 1u << 20, 512ull, 1, 5, 3);
#endif
    for (int ch = 0; ch < 4; ++ch) {
        const int cur = ch & 1;
        unsigned short* qsc = (unsigned short*)(smem + cur * 17408);
        unsigned short* ksc = (unsigned short*)(smem + cur * 17408 + 8704);
#if HAVE_TDM
        unsigned char* nx = smem + (cur ^ 1) * 17408;
        if (ch < 3) {
            if (wvu == 0)
                tdm_load_tile_2d(nx, &QT[colbase * 512 + (ch + 1) * 128], 128, 32, 512,
                                 1u << 20, 512ull, 1, 5, 3);
            if (wvu == 1)
                tdm_load_tile_2d(nx + 8704, &KT[colbase * 512 + (ch + 1) * 128], 128, 32, 512,
                                 1u << 20, 512ull, 1, 5, 3);
            if (wvu < 2) __builtin_amdgcn_s_wait_tensorcnt(1);
        } else {
            if (wvu < 2) __builtin_amdgcn_s_wait_tensorcnt(0);
        }
        __syncthreads();
#else
        __syncthreads();
        #pragma unroll
        for (int i = 0; i < 2; ++i) {
            const int idx = tid + i * 256;     // 512 chunks of 8
            const int t = idx >> 4, cc = (idx & 15) * 8;
            *(u32x4*)&qsc[t * 136 + cc] = *(const u32x4*)&QT[(colbase + t) * 512 + ch * 128 + cc];
            *(u32x4*)&ksc[t * 136 + cc] = *(const u32x4*)&KT[(colbase + t) * 512 + ch * 128 + cc];
        }
        __syncthreads();
#endif
        #pragma unroll
        for (int j = 0; j < 4; ++j) {
            if (((ch * 4 + j) & 7) != wv) continue;
            const int c0 = j * 32;
            Frag a[2], bb[2];
            #pragma unroll
            for (int mt = 0; mt < 2; ++mt) {
                a[mt].q[0] = *(const u32x4*)&qsc[(mt * 16 + lrow) * 136 + c0 + halfsel * 8];
                a[mt].q[1] = *(const u32x4*)&qsc[(mt * 16 + lrow) * 136 + c0 + halfsel * 8 + 16];
            }
            #pragma unroll
            for (int nt = 0; nt < 2; ++nt) {
                bb[nt].q[0] = *(const u32x4*)&ksc[(nt * 16 + lrow) * 136 + c0 + halfsel * 8];
                bb[nt].q[1] = *(const u32x4*)&ksc[(nt * 16 + lrow) * 136 + c0 + halfsel * 8 + 16];
            }
            #pragma unroll
            for (int mt = 0; mt < 2; ++mt)
                #pragma unroll
                for (int nt = 0; nt < 2; ++nt)
                    Sacc[mt][nt] = __builtin_amdgcn_wmma_f32_16x16x32_bf16(
                        false, a[mt].v, false, bb[nt].v, (short)0, Sacc[mt][nt], false, false);
        }
        __syncthreads();
    }
    // partial-S reduction across waves
    #pragma unroll
    for (int mt = 0; mt < 2; ++mt)
        #pragma unroll
        for (int nt = 0; nt < 2; ++nt)
            #pragma unroll
            for (int r = 0; r < 8; ++r)
                sred[(wv * 32 + mt * 16 + r + halfsel * 8) * 32 + nt * 16 + lrow] = Sacc[mt][nt][r];
    __syncthreads();
    for (int e = tid; e < 1024; e += 256) {
        float tot = 0.f;
        #pragma unroll
        for (int w = 0; w < 8; ++w) tot += sred[w * 1024 + e];
        sred[e] = tot;
    }
    __syncthreads();
    // ---- causal softmax (wave 0, one lane per row) ----
    if (tid < 32) {
        const int t = tid;
        const float scale = 0.0441941738241592f;   // 1/sqrt(512)
        float m = -1e30f;
        for (int s = 0; s <= t; ++s) m = fmaxf(m, sred[t * 32 + s] * scale);
        float sum = 0.f;
        float ev[32];
        #pragma unroll
        for (int s = 0; s < 32; ++s) {
            const float v = (s <= t) ? __expf(sred[t * 32 + s] * scale - m) : 0.f;
            ev[s] = v; sum += v;
        }
        const float inv = 1.f / sum;
        #pragma unroll
        for (int s = 0; s < 16; ++s)
            ((unsigned*)&pbf[t * 40])[s] = pack_bf2(ev[2 * s] * inv, ev[2 * s + 1] * inv);
    }
    __syncthreads();
    // ---- O-phase: double-buffered v chunk pipeline; OT[t][c] = sum_s P[t,s]*V[c,s] ----
#if HAVE_TDM
    if (wvu == 0)   // v set 0 aliases dead q/k region
        tdm_load_tile_2d(smem, &Vb[colbase], 32, 128, (unsigned)N_COLS, 512,
                         (unsigned long long)N_COLS, 1, 3, 3);
#endif
    for (int ch = 0; ch < 4; ++ch) {
        const int cur = ch & 1;
        unsigned short* vsc = (unsigned short*)(smem + cur * 10240);  // [128][40] padded
#if HAVE_TDM
        if (ch < 3) {
            if (wvu == 0)
                tdm_load_tile_2d(smem + (cur ^ 1) * 10240,
                                 &Vb[(size_t)((ch + 1) * 128) * N_COLS + colbase],
                                 32, 128, (unsigned)N_COLS, 512,
                                 (unsigned long long)N_COLS, 1, 3, 3);
            if (wvu == 0) __builtin_amdgcn_s_wait_tensorcnt(1);
        } else {
            if (wvu == 0) __builtin_amdgcn_s_wait_tensorcnt(0);
        }
        __syncthreads();
#else
        __syncthreads();
        #pragma unroll
        for (int i = 0; i < 2; ++i) {
            const int idx = tid + i * 256;
            const int cl = idx >> 2, s0 = (idx & 3) * 8;
            *(u32x4*)&vsc[cl * 40 + s0] =
                *(const u32x4*)&Vb[(size_t)(ch * 128 + cl) * N_COLS + colbase + s0];
        }
        __syncthreads();
#endif
        Frag a, bb[2];
        const int crow = wv * 16 + lrow;
        a.q[0] = *(const u32x4*)&vsc[crow * 40 + halfsel * 8];
        a.q[1] = *(const u32x4*)&vsc[crow * 40 + halfsel * 8 + 16];
        #pragma unroll
        for (int nt = 0; nt < 2; ++nt) {
            bb[nt].q[0] = *(const u32x4*)&pbf[(nt * 16 + lrow) * 40 + halfsel * 8];
            bb[nt].q[1] = *(const u32x4*)&pbf[(nt * 16 + lrow) * 40 + halfsel * 8 + 16];
            const v8f o = __builtin_amdgcn_wmma_f32_16x16x32_bf16(
                false, a.v, false, bb[nt].v, (short)0, zero, false, false);
            // lane holds column t; its 8 accumulator rows are consecutive channels -> b128 store
            u32x4 pk;
            #pragma unroll
            for (int j = 0; j < 4; ++j) pk[j] = pack_bf2(o[2 * j], o[2 * j + 1]);
            const size_t n = colbase + (size_t)(nt * 16 + lrow);
            *(u32x4*)&OT[n * 512 + ch * 128 + wv * 16 + halfsel * 8] = pk;
        }
        __syncthreads();
    }
}

// ------- kernel 6: transpose back + residual add (f32 out); Pj is [o][n] -------
__global__ void __launch_bounds__(256) residual_kernel(const float* __restrict__ x,
                                                       const unsigned short* __restrict__ Pj,
                                                       float* __restrict__ out) {
    const int blk = blockIdx.x;
    const int y = blk & 31;
    const int c = (blk >> 5) & 511;
    const int b = blk >> 14;
    __shared__ float tile[32][36];   // [w][t], 144B rows
    const size_t rowbase = (size_t)c * N_COLS + ((size_t)b * HW + (size_t)y * 32) * T_DIM;
    const int e = threadIdx.x * 4;   // 1024 elems, 4 per thread
    {
        const int w = e >> 5, t0 = e & 31;
        const u32x2 pv = *(const u32x2*)&Pj[rowbase + e];
        f32x4 f;
        f[0] = bf2f(pv[0] & 0xFFFFu); f[1] = bf2f(pv[0] >> 16);
        f[2] = bf2f(pv[1] & 0xFFFFu); f[3] = bf2f(pv[1] >> 16);
        *(f32x4*)&tile[w][t0] = f;
    }
    __syncthreads();
    {
        const int t = e >> 5, w0 = e & 31;
        const size_t xi = ((((size_t)b * C_DIM + c) * T_DIM + t) * 32 + y) * 32 + w0;
        const f32x4 xv = *(const f32x4*)&x[xi];
        f32x4 o;
        #pragma unroll
        for (int j = 0; j < 4; ++j) o[j] = xv[j] + tile[w0 + j][t];
        *(f32x4*)&out[xi] = o;
    }
}

// ---------------- launch ----------------
extern "C" void kernel_launch(void* const* d_in, const int* in_sizes, int n_in,
                              void* d_out, int out_size, void* d_ws, size_t ws_size,
                              hipStream_t stream) {
    (void)in_sizes; (void)n_in; (void)out_size; (void)ws_size;
    const float* x     = (const float*)d_in[0];
    const float* gamma = (const float*)d_in[1];
    const float* beta  = (const float*)d_in[2];
    const float* wq    = (const float*)d_in[3];
    const float* wk    = (const float*)d_in[4];
    const float* wvv   = (const float*)d_in[5];
    const float* wproj = (const float*)d_in[6];
    float* out = (float*)d_out;

    char* ws = (char*)d_ws;
    float* stats = (float*)ws;
    size_t off = 4096;
    unsigned short* Wbf = (unsigned short*)(ws + off); off += (size_t)4 * 262144 * 2;  // 2MB
    const size_t BUF = (size_t)C_DIM * N_COLS * sizeof(unsigned short);                // 128MB
    unsigned short* bufA = (unsigned short*)(ws + off); off += BUF;  // HT, then OT
    unsigned short* bufB = (unsigned short*)(ws + off); off += BUF;  // QT, then Pj
    unsigned short* bufC = (unsigned short*)(ws + off); off += BUF;  // KT
    unsigned short* bufD = (unsigned short*)(ws + off); off += BUF;  // V

    convert_w_kernel<<<1024, 256, 0, stream>>>(wq, wk, wvv, wproj, Wbf);
    gn_stats_kernel<<<B_DIM * T_DIM, 256, 0, stream>>>(x, stats);
    norm_ht_kernel<<<B_DIM * 32 * 32 * 8, 256, 0, stream>>>(x, gamma, beta, stats, bufA);

    // QT[n][o] = sum_c HT[n,c]*Wq[o,c]   (M = N_COLS rows, N = 512)
    dim3 gq(C_DIM / 64, N_COLS / 128);
    gemm_nt_kernel<<<gq, 256, 0, stream>>>(bufA, Wbf + 0 * 262144, bufB, (size_t)C_DIM);
    gemm_nt_kernel<<<gq, 256, 0, stream>>>(bufA, Wbf + 1 * 262144, bufC, (size_t)C_DIM);
    // V[o][n] = sum_c Wv[o,c]*HT[n,c]    (M = 512 rows, N = N_COLS)
    dim3 gv(N_COLS / 64, C_DIM / 128);
    gemm_nt_kernel<<<gv, 256, 0, stream>>>(Wbf + 2 * 262144, bufA, bufD, (size_t)N_COLS);

    attention_kernel<<<P_DIM, 256, 0, stream>>>(bufB, bufC, bufD, bufA);  // -> OT in bufA

    // Pj[o][n] = sum_c Wproj[o,c]*OT[n,c]
    gemm_nt_kernel<<<gv, 256, 0, stream>>>(Wbf + 3 * 262144, bufA, bufB, (size_t)N_COLS);
    residual_kernel<<<B_DIM * C_DIM * 32, 256, 0, stream>>>(x, bufB, out);
}
